// RelationAwareAttention_26027501814593
// MI455X (gfx1250) — matrix operand order repaired and checked by reading, not compiled, not run on hardware
//
#include <hip/hip_runtime.h>
#include <hip/hip_bf16.h>

// ---------------------------------------------------------------------------
// Types for CDNA5 WMMA (wave32, v_wmma_f32_16x16x32_bf16) and TDM
// ---------------------------------------------------------------------------
typedef __attribute__((ext_vector_type(16))) __bf16 v16bf;
typedef __attribute__((ext_vector_type(8)))  __bf16 v8bf;
typedef __attribute__((ext_vector_type(8)))  float  v8f;
typedef __attribute__((ext_vector_type(4))) unsigned int u32x4;
typedef __attribute__((ext_vector_type(8))) int         i32x8;
typedef __attribute__((ext_vector_type(4))) int         i32x4;

#define EMBED 512
#define SEQ   512
#define BATCH 8
#define HEADS 8
#define DPH   64
#define NREL  3
#define MROWS (BATCH * SEQ)   // 4096

__device__ __forceinline__ v8f wmma_bf16(v16bf a, v16bf b, v8f c) {
    return __builtin_amdgcn_wmma_f32_16x16x32_bf16(
        false, a, false, b, (short)0, c, false, false);
}

// A fragment: 16(M) x 32(K) bf16 from row-major LDS tile.
__device__ __forceinline__ v16bf load_a_bf16(const __bf16* base, int ld, int lane) {
    const int m  = lane & 15;
    const int kb = (lane >> 4) * 8;
    const __bf16* p = base + m * ld + kb;
    v8bf lo = *(const v8bf*)(p);
    v8bf hi = *(const v8bf*)(p + 16);
    v16bf a;
#pragma unroll
    for (int i = 0; i < 8; ++i) { a[i] = lo[i]; a[i + 8] = hi[i]; }
    return a;
}

// B fragment: 32(K) x 16(N) bf16 from an [N][K] (K-contiguous) LDS tile.
__device__ __forceinline__ v16bf load_b_bf16(const __bf16* base, int ld, int lane) {
    const int n  = lane & 15;
    const int kb = (lane >> 4) * 8;
    const __bf16* p = base + n * ld + kb;
    v8bf lo = *(const v8bf*)(p);
    v8bf hi = *(const v8bf*)(p + 16);
    v16bf b;
#pragma unroll
    for (int i = 0; i < 8; ++i) { b[i] = lo[i]; b[i + 8] = hi[i]; }
    return b;
}

__device__ __forceinline__ void st4_bf16(__bf16* d, float4 f) {
    d[0] = (__bf16)f.x; d[1] = (__bf16)f.y; d[2] = (__bf16)f.z; d[3] = (__bf16)f.w;
}

// ---------------------------------------------------------------------------
// Tensor Data Mover: 2-D rectangle copy global(bf16) -> LDS, with HW padding
// that produces an 80-element (160B) LDS row stride from 64-element rows.
// D# packing per CDNA5 ISA.
// ---------------------------------------------------------------------------
#define KVLD 80            // bf16 LDS row stride (160B: 128B data + 32B pad)
#define PAD_INTERVAL_C 4   // pad after 1<<4 8-byte units = 128B
#define PAD_AMOUNT_C   7   // pad by (7+1) DWORDs = 32B

__device__ __forceinline__ unsigned lds_addr_of(const void* p) {
    return (unsigned)(unsigned long long)p;   // low 32 bits = LDS byte address
}

__device__ __forceinline__ void tdm_load_tile_bf16(unsigned lds_addr, const __bf16* gsrc,
                                                   unsigned rows, unsigned row_elems,
                                                   unsigned row_stride_elems) {
    const unsigned long long ga = (unsigned long long)gsrc;
    u32x4 g0;
    g0[0] = 1u;                                           // count=1, user mode
    g0[1] = lds_addr;                                     // lds_addr [63:32]
    g0[2] = (unsigned)(ga & 0xFFFFFFFFu);                 // global_addr lo
    g0[3] = (unsigned)((ga >> 32) & 0x01FFFFFFu) | (2u << 30);  // addr hi | type=2

    const unsigned td0 = row_stride_elems;                // tensor_dim0 (elems)
    const unsigned td1 = rows;                            // tensor_dim1
    i32x8 g1;
    g1[0] = (int)((1u << 16) |                            // data_size=1 -> 2 bytes
                  (1u << 20) |                            // pad_enable
                  ((unsigned)PAD_INTERVAL_C << 22) |
                  ((unsigned)PAD_AMOUNT_C << 25));
    g1[1] = (int)((td0 & 0xFFFFu) << 16);                 // bits 63:48 = dim0[15:0]
    g1[2] = (int)(((td0 >> 16) & 0xFFFFu) | ((td1 & 0xFFFFu) << 16));
    g1[3] = (int)(((td1 >> 16) & 0xFFFFu) | ((row_elems & 0xFFFFu) << 16)); // tile_dim0
    g1[4] = (int)(rows & 0xFFFFu);                        // tile_dim1 (tile_dim2=0)
    g1[5] = (int)row_stride_elems;                        // tensor_dim0_stride lo32
    g1[6] = 0;                                            // stride hi16 | dim1_stride
    g1[7] = 0;
    const i32x4 z4 = {0, 0, 0, 0};
#if defined(__clang_major__) && (__clang_major__ >= 23)
    const i32x8 z8 = {0, 0, 0, 0, 0, 0, 0, 0};
    __builtin_amdgcn_tensor_load_to_lds(g0, g1, z4, z4, z8, 0);
#else
    __builtin_amdgcn_tensor_load_to_lds(g0, g1, z4, z4, 0);
#endif
}

// ---------------------------------------------------------------------------
// GEMM + bias:  C[4096,512] = A(f32) @ W(f32) + bias, bf16 side-copies:
//   bf_rm: row-major bf16 copy of C (for q,k)
//   bf_vt: per-(b,h) transposed bf16 copy vT[b][h][d][t] (for v -> B-fragments)
// ---------------------------------------------------------------------------
#define LDT 40   // bf16 element stride of GEMM LDS tiles (80B rows)

__global__ __launch_bounds__(256)
void gemm_bias_kernel(const float* __restrict__ A, const float* __restrict__ W,
                      const float* __restrict__ bias, float* __restrict__ C,
                      __bf16* __restrict__ bf_rm, __bf16* __restrict__ bf_vt) {
    __shared__ __bf16 sA[128 * LDT];   // [m][k] row-major, 128x32
    __shared__ __bf16 sB[128 * LDT];   // [n][k] (W tile transposed), 128x32

    const int tid  = threadIdx.x;
    const int wave = tid >> 5;
    const int lane = tid & 31;
    const int m0 = blockIdx.x * 128;
    const int n0 = blockIdx.y * 128;
    const int wr = (wave & 3) * 32;
    const int wc = (wave >> 2) * 64;

    v8f acc[2][4];
#pragma unroll
    for (int i = 0; i < 2; ++i)
#pragma unroll
        for (int j = 0; j < 4; ++j)
#pragma unroll
            for (int e = 0; e < 8; ++e) acc[i][j][e] = 0.0f;

    for (int kk = 0; kk < EMBED; kk += 32) {
        // Load phase: all 8 global float4s into registers first so the
        // compiler can clause the loads and drain with partial waits.
        float4 fa[4], fw[4];
#pragma unroll
        for (int i = 0; i < 4; ++i) {
            const int lin = tid + 256 * i;
            const int r = lin >> 3, c4 = lin & 7;
            fa[i] = *(const float4*)(A + (size_t)(m0 + r) * EMBED + kk + c4 * 4);
        }
#pragma unroll
        for (int i = 0; i < 4; ++i) {
            const int lin = tid + 256 * i;
            const int r = lin >> 5, c4 = lin & 31;
            fw[i] = *(const float4*)(W + (size_t)(kk + r) * EMBED + n0 + c4 * 4);
        }
        if (kk + 32 < EMBED) {
            __builtin_prefetch(A + (size_t)(m0 + (tid >> 1)) * EMBED + kk + 32, 0, 1);
            __builtin_prefetch(W + (size_t)(kk + 32 + (tid >> 3)) * EMBED + n0, 0, 1);
        }
        // Convert + store phase.
#pragma unroll
        for (int i = 0; i < 4; ++i) {
            const int lin = tid + 256 * i;
            const int r = lin >> 3, c4 = lin & 7;
            st4_bf16(&sA[r * LDT + c4 * 4], fa[i]);
        }
#pragma unroll
        for (int i = 0; i < 4; ++i) {
            const int lin = tid + 256 * i;
            const int r = lin >> 5, c4 = lin & 31;
            sB[(c4 * 4 + 0) * LDT + r] = (__bf16)fw[i].x;
            sB[(c4 * 4 + 1) * LDT + r] = (__bf16)fw[i].y;
            sB[(c4 * 4 + 2) * LDT + r] = (__bf16)fw[i].z;
            sB[(c4 * 4 + 3) * LDT + r] = (__bf16)fw[i].w;
        }
        __syncthreads();

        v16bf af[2], bf[4];
#pragma unroll
        for (int i = 0; i < 2; ++i) af[i] = load_a_bf16(&sA[(wr + 16 * i) * LDT], LDT, lane);
#pragma unroll
        for (int j = 0; j < 4; ++j) bf[j] = load_b_bf16(&sB[(wc + 16 * j) * LDT], LDT, lane);
#pragma unroll
        for (int i = 0; i < 2; ++i)
#pragma unroll
            for (int j = 0; j < 4; ++j) acc[i][j] = wmma_bf16(af[i], bf[j], acc[i][j]);
        __syncthreads();
    }

    const int hi = lane >> 4, lo = lane & 15;
#pragma unroll
    for (int i = 0; i < 2; ++i)
#pragma unroll
        for (int j = 0; j < 4; ++j) {
            const int col = n0 + wc + 16 * j + lo;
            const float bv = bias[col];
#pragma unroll
            for (int jj = 0; jj < 8; ++jj) {
                const int row = m0 + wr + 16 * i + jj + 8 * hi;
                const float val = acc[i][j][jj] + bv;
                C[(size_t)row * EMBED + col] = val;
                if (bf_rm) bf_rm[(size_t)row * EMBED + col] = (__bf16)val;
                if (bf_vt) {
                    const int bb = row >> 9, t = row & (SEQ - 1);
                    const int hh = col >> 6, dd = col & (DPH - 1);
                    bf_vt[((size_t)(bb * HEADS + hh) * DPH + dd) * SEQ + t] = (__bf16)val;
                }
            }
        }
}

// ---------------------------------------------------------------------------
// qe[b,s,h,r] = q[b,s,h*64:..] . rel_k[r]   (rank-3 factorization of QE)
// ---------------------------------------------------------------------------
__global__ __launch_bounds__(256)
void qe_kernel(const float* __restrict__ q, const float* __restrict__ rel_k,
               float* __restrict__ qe) {
    const int idx = blockIdx.x * 256 + threadIdx.x;
    if (idx >= BATCH * SEQ * HEADS * NREL) return;
    const int r  = idx % NREL;
    const int h  = (idx / NREL) % HEADS;
    const int bs = idx / (NREL * HEADS);
    const float* qp = q + (size_t)bs * EMBED + h * DPH;
    const float* rp = rel_k + r * DPH;
    float acc = 0.0f;
#pragma unroll 8
    for (int d = 0; d < DPH; ++d) acc += qp[d] * rp[d];
    qe[idx] = acc;
}

// ---------------------------------------------------------------------------
// Attention: block per (b, h, 64-row q tile). 64x512 score block in LDS
// (320 KB/WGP). q/k/v tiles staged by TDM (double-buffered, HW-padded),
// WMMA scores -> softmax with QE gather + relation masses -> WMMA e@v.
// ---------------------------------------------------------------------------
#define ATTN_SMEM_BYTES (64*512*4 + 3*64*KVLD*2 + 4*64*4 + 4*64*3*4 + 64*4 + 64*4)

__global__ __launch_bounds__(256)
void attn_kernel(const __bf16* __restrict__ qbf, const __bf16* __restrict__ kbf,
                 const __bf16* __restrict__ vtb, const float* __restrict__ qe,
                 const int* __restrict__ rels, const float* __restrict__ rel_v,
                 float* __restrict__ ctx) {
    extern __shared__ unsigned char smem_raw[];
    float*  sc     = (float*)smem_raw;                       // 64 x 512
    __bf16* qtile  = (__bf16*)(smem_raw + 64 * 512 * 4);     // 64 x KVLD
    __bf16* kv0    = qtile + 64 * KVLD;                      // 64 x KVLD
    __bf16* kv1    = kv0 + 64 * KVLD;                        // 64 x KVLD
    float*  red    = (float*)(kv1 + 64 * KVLD);              // 4 x 64
    float*  wrp    = red + 4 * 64;                           // 4 x 64 x 3
    float*  rowmax = wrp + 4 * 64 * 3;                       // 64
    float*  rowsum = rowmax + 64;                            // 64

    const int qt = blockIdx.x, h = blockIdx.y, b = blockIdx.z;
    const int s0 = qt * 64;
    const int tid  = threadIdx.x;
    const int wave = tid >> 5;
    const int lane = tid & 31;
    const int mi = wave & 3;
    const int nc = (wave >> 2) * 32;

    const __bf16* kbase  = kbf + ((size_t)(b * SEQ) * EMBED + h * DPH);
    const __bf16* vtbase = vtb + (size_t)(b * HEADS + h) * DPH * SEQ;

    // ---- Phase 1: TDM-stage q tile; prefetch k tile 0 behind it ----
    if (wave == 0) {
        tdm_load_tile_bf16(lds_addr_of(qtile),
                           qbf + ((size_t)(b * SEQ + s0) * EMBED + h * DPH),
                           64, 64, EMBED);
        tdm_load_tile_bf16(lds_addr_of(kv0), kbase, 64, 64, EMBED);
        __builtin_amdgcn_s_wait_tensorcnt(1);   // q landed; k tile 0 in flight
    }
    __syncthreads();

    v16bf aq[2];
    aq[0] = load_a_bf16(&qtile[(16 * mi) * KVLD + 0],  KVLD, lane);
    aq[1] = load_a_bf16(&qtile[(16 * mi) * KVLD + 32], KVLD, lane);

    // ---- Phase 2: raw QK scores via WMMA, TDM double-buffered k tiles ----
    for (int kt = 0; kt < 8; ++kt) {
        __bf16* cur = (kt & 1) ? kv1 : kv0;
        __bf16* nxt = (kt & 1) ? kv0 : kv1;
        if (wave == 0) {
            if (kt < 7) {
                tdm_load_tile_bf16(lds_addr_of(nxt),
                                   kbase + (size_t)(kt + 1) * 64 * EMBED,
                                   64, 64, EMBED);
                __builtin_amdgcn_s_wait_tensorcnt(1);  // tile kt done, kt+1 in flight
            } else {
                __builtin_amdgcn_s_wait_tensorcnt(0);
            }
        }
        __syncthreads();

        const int t0 = kt * 64;
#pragma unroll
        for (int tt = 0; tt < 2; ++tt) {
            v8f c;
#pragma unroll
            for (int e = 0; e < 8; ++e) c[e] = 0.0f;
#pragma unroll
            for (int hk = 0; hk < 2; ++hk) {
                v16bf bkf = load_b_bf16(&cur[(nc + 16 * tt) * KVLD + hk * 32], KVLD, lane);
                c = wmma_bf16(aq[hk], bkf, c);
            }
            const int colb = t0 + nc + 16 * tt + (lane & 15);
            const int rowb = 16 * mi + 8 * (lane >> 4);
#pragma unroll
            for (int jj = 0; jj < 8; ++jj) sc[(rowb + jj) * 512 + colb] = c[jj];
        }
        __syncthreads();
    }

    // ---- Phase 3: add QE (gathered), scale, softmax masses ----
    {
        const int row = tid >> 2, seg = tid & 3;
        const int* relrow = rels + ((size_t)(b * SEQ) + s0 + row) * SEQ + seg * 128;
        const float* qerow = qe + ((size_t)((b * SEQ + s0 + row) * HEADS + h)) * NREL;
        float* scrow = sc + row * 512 + seg * 128;
        const float scale = 0.125f;  // DPH^-0.5

        float lmax = -1e30f;
        for (int c = 0; c < 128; ++c) {
            const int rl = relrow[c];
            const float v = (scrow[c] + qerow[rl]) * scale;
            scrow[c] = v;
            lmax = fmaxf(lmax, v);
        }
        red[seg * 64 + row] = lmax;
        __syncthreads();
        if (seg == 0) {
            float m = red[row];
            m = fmaxf(m, red[64 + row]);
            m = fmaxf(m, red[128 + row]);
            m = fmaxf(m, red[192 + row]);
            rowmax[row] = m;
        }
        __syncthreads();

        const float m = rowmax[row];
        float lsum = 0.0f, lw0 = 0.0f, lw1 = 0.0f, lw2 = 0.0f;
        for (int c = 0; c < 128; ++c) {
            const int rl = relrow[c];
            const float e = __expf(scrow[c] - m);
            scrow[c] = e;
            lsum += e;
            if (rl == 0) lw0 += e; else if (rl == 1) lw1 += e; else lw2 += e;
        }
        red[seg * 64 + row] = lsum;
        wrp[(seg * 64 + row) * 3 + 0] = lw0;
        wrp[(seg * 64 + row) * 3 + 1] = lw1;
        wrp[(seg * 64 + row) * 3 + 2] = lw2;
        __syncthreads();
        if (seg == 0) {
            rowsum[row] = red[row] + red[64 + row] + red[128 + row] + red[192 + row];
            for (int k = 0; k < 3; ++k)
                wrp[row * 3 + k] = wrp[row * 3 + k] + wrp[(64 + row) * 3 + k] +
                                   wrp[(128 + row) * 3 + k] + wrp[(192 + row) * 3 + k];
        }
        __syncthreads();
    }

    // ---- Phase 4: O = e @ v via WMMA, TDM double-buffered vT tiles ----
    v8f oacc[2];
#pragma unroll
    for (int tt = 0; tt < 2; ++tt)
#pragma unroll
        for (int e = 0; e < 8; ++e) oacc[tt][e] = 0.0f;

    if (wave == 0)
        tdm_load_tile_bf16(lds_addr_of(kv0), vtbase, 64, 64, SEQ);

    for (int vt = 0; vt < 8; ++vt) {
        __bf16* cur = (vt & 1) ? kv1 : kv0;
        __bf16* nxt = (vt & 1) ? kv0 : kv1;
        if (wave == 0) {
            if (vt < 7) {
                tdm_load_tile_bf16(lds_addr_of(nxt), vtbase + (vt + 1) * 64, 64, 64, SEQ);
                __builtin_amdgcn_s_wait_tensorcnt(1);
            } else {
                __builtin_amdgcn_s_wait_tensorcnt(0);
            }
        }
        __syncthreads();

        const int t0 = vt * 64;
#pragma unroll
        for (int hk = 0; hk < 2; ++hk) {
            const int ml = 16 * mi + (lane & 15);
            const int kbse = (lane >> 4) * 8;
            const float* p = sc + ml * 512 + t0 + hk * 32 + kbse;
            v16bf ae;
#pragma unroll
            for (int e = 0; e < 8; ++e) { ae[e] = (__bf16)p[e]; ae[e + 8] = (__bf16)p[e + 16]; }
#pragma unroll
            for (int tt = 0; tt < 2; ++tt) {
                v16bf bvf = load_b_bf16(&cur[(nc + 16 * tt) * KVLD + hk * 32], KVLD, lane);
                oacc[tt] = wmma_bf16(ae, bvf, oacc[tt]);
            }
        }
        __syncthreads();
    }

    // ---- Phase 5: ctx = (O + sum_r w_r * rel_v[r]) / rowsum ----
    {
        const int hi = lane >> 4, lo = lane & 15;
#pragma unroll
        for (int tt = 0; tt < 2; ++tt) {
            const int d = nc + 16 * tt + lo;
#pragma unroll
            for (int jj = 0; jj < 8; ++jj) {
                const int rloc = 16 * mi + jj + 8 * hi;
                const float inv = 1.0f / rowsum[rloc];
                const float ee = wrp[rloc * 3 + 0] * rel_v[0 * DPH + d] +
                                 wrp[rloc * 3 + 1] * rel_v[1 * DPH + d] +
                                 wrp[rloc * 3 + 2] * rel_v[2 * DPH + d];
                ctx[(size_t)(b * SEQ + s0 + rloc) * EMBED + h * DPH + d] = (oacc[tt][jj] + ee) * inv;
            }
        }
    }
}

// ---------------------------------------------------------------------------
// Launch
// ---------------------------------------------------------------------------
extern "C" void kernel_launch(void* const* d_in, const int* in_sizes, int n_in,
                              void* d_out, int out_size, void* d_ws, size_t ws_size,
                              hipStream_t stream) {
    const float* query = (const float*)d_in[0];
    const float* key   = (const float*)d_in[1];
    const float* value = (const float*)d_in[2];
    const int*   rels  = (const int*)d_in[3];
    const float* Wq = (const float*)d_in[4];
    const float* bq = (const float*)d_in[5];
    const float* Wk = (const float*)d_in[6];
    const float* bk = (const float*)d_in[7];
    const float* Wv = (const float*)d_in[8];
    const float* bv = (const float*)d_in[9];
    const float* Wo = (const float*)d_in[10];
    const float* bo = (const float*)d_in[11];
    const float* rel_k = (const float*)d_in[12];
    const float* rel_v = (const float*)d_in[13];
    float* out = (float*)d_out;

    // Workspace: 4 f32 buffers (4096x512) + qe + 3 bf16 buffers (~46 MB)
    float* qbuf = (float*)d_ws;
    float* kbuf = qbuf + (size_t)MROWS * EMBED;
    float* vbuf = kbuf + (size_t)MROWS * EMBED;
    float* cbuf = vbuf + (size_t)MROWS * EMBED;
    float* qebf = cbuf + (size_t)MROWS * EMBED;
    __bf16* qbf = (__bf16*)(qebf + (size_t)BATCH * SEQ * HEADS * NREL);
    __bf16* kbf = qbf + (size_t)MROWS * EMBED;
    __bf16* vtb = kbf + (size_t)MROWS * EMBED;

    (void)in_sizes; (void)n_in; (void)out_size; (void)ws_size;

    (void)hipFuncSetAttribute((const void*)attn_kernel,
                              hipFuncAttributeMaxDynamicSharedMemorySize,
                              (int)ATTN_SMEM_BYTES);

    dim3 gg(MROWS / 128, EMBED / 128, 1), gb(256);
    gemm_bias_kernel<<<gg, gb, 0, stream>>>(query, Wq, bq, qbuf, qbf, nullptr);
    gemm_bias_kernel<<<gg, gb, 0, stream>>>(key,   Wk, bk, kbuf, kbf, nullptr);
    gemm_bias_kernel<<<gg, gb, 0, stream>>>(value, Wv, bv, vbuf, nullptr, vtb);

    qe_kernel<<<(BATCH * SEQ * HEADS * NREL + 255) / 256, 256, 0, stream>>>(qbuf, rel_k, qebf);

    attn_kernel<<<dim3(SEQ / 64, HEADS, BATCH), 256, ATTN_SMEM_BYTES, stream>>>(
        qbf, kbf, vtb, qebf, rels, rel_v, cbuf);

    gemm_bias_kernel<<<gg, gb, 0, stream>>>(cbuf, Wo, bo, out, nullptr, nullptr);
}
